// NeuralNet_21921513079546
// MI455X (gfx1250) — matrix operand order repaired and checked by reading, not compiled, not run on hardware
//
#include <hip/hip_runtime.h>
#include <hip/hip_bf16.h>

typedef __attribute__((ext_vector_type(16))) _Float16 v16h;
typedef __attribute__((ext_vector_type(8)))  _Float16 v8h;
typedef __attribute__((ext_vector_type(8)))  float    v8f;

#define PI_F 3.14159265358979323846f

// A-fragment element -> K mapping (16-bit A, 16x32):  e<8 -> hl*8+e ; e>=8 -> 16+hl*8+(e-8)
// B-fragment element -> K mapping (16-bit B, 32x16):  same as A: e<8 -> hl*8+e ; e>=8 -> 16+hl*8+(e-8)
__device__ __forceinline__ int kmapA(int e, int hl) {
    return (e < 8) ? (hl * 8 + e) : (16 + hl * 8 + (e - 8));
}

// ---------------------------------------------------------------------------
// Kernel 1: fully fused per-image pipeline (one workgroup = one image).
//   stage 1: conv1 (valid 28->26, 1->32ch) via WMMA in TRANSPOSED form:
//            A = W1 (two 16-channel M-tiles, K = 9 taps padded to 32, true
//            zeros), B = im2col of x (unconditional clamped loads; elements
//            at k>=9 are finite garbage multiplied by A's zeros).
//            D = [channel x position] -> two ds_store_b128 per tile.
//   stage 2: conv2 (valid 26->24, 32->3ch) via 9x WMMA, A frags preloaded
//   stage 3: SIFT-128, deterministic fixed-order partial sums (144 threads)
//   output : feats f16 [img][384] to global workspace
// ---------------------------------------------------------------------------
__global__ __launch_bounds__(256)
void fused_conv_sift_kernel(const float* __restrict__ x,
                            const float* __restrict__ W1,
                            const float* __restrict__ b1,
                            const float* __restrict__ W2,
                            const float* __restrict__ b2,
                            _Float16* __restrict__ feats)
{
    __shared__ _Float16 s_xh[784];                   //  1568 B  input as f16
    __shared__ _Float16 s_w1h[288];                  //  [ch][tap] f16
    __shared__ float    s_b1[32];
    __shared__ _Float16 s_w2h[864];                  //  [c][ci][tap] f16
    __shared__ float    s_b2[3];
    __shared__ __align__(16) _Float16 s_r1[21632];   //  [pix=y*26+x][ci], 43264 B
    __shared__ float    s_c2[3 * 576];               //  [c][24*24], 6912 B
    __shared__ float    s_part[3 * 16 * 3 * 8];      //  SIFT partials, 4608 B
    __shared__ float    s_hist[3 * 128];             //  1536 B
    __shared__ float    s_trig[16];                  //  cos[8], sin[8]
    __shared__ float    s_scale[3];

    const int tid  = threadIdx.x;
    const int img  = blockIdx.x;
    const int wv   = tid >> 5;
    const int lane = tid & 31;
    const int mn   = lane & 15;
    const int hl   = lane >> 4;

    // ---- stage 0: cooperative loads / conversions -----------------------
    for (int i = tid; i < 784; i += 256) s_xh[i]  = (_Float16)x[img * 784 + i];
    for (int i = tid; i < 288; i += 256) s_w1h[i] = (_Float16)W1[i];   // [ch][u][v] flat
    if (tid < 32) s_b1[tid] = b1[tid];
    for (int i = tid; i < 864; i += 256) s_w2h[i] = (_Float16)W2[i];   // (c*32+ci)*9+tap
    if (tid < 3)  s_b2[tid] = b2[tid];
    if (tid < 8) {
        float a = PI_F / 8.0f + (float)tid * (PI_F / 4.0f);
        s_trig[tid]     = cosf(a);
        s_trig[8 + tid] = sinf(a);
    }
    __syncthreads();

    // ---- stage 1: conv1 + ReLU via WMMA (transposed: D = ch x pos) ------
    {
        // A fragments = weights: m = channel-in-tile, K = taps (zeros at k>=9)
        v16h aw0, aw1;
        #pragma unroll
        for (int e = 0; e < 16; ++e) {
            int k  = kmapA(e, hl);
            int kc = (k < 9) ? k : 8;                 // clamped, always-safe load
            _Float16 w0 = s_w1h[mn * 9 + kc];
            _Float16 w1v = s_w1h[(16 + mn) * 9 + kc];
            aw0[e] = (k < 9) ? w0  : (_Float16)0.0f;  // register select, no branch
            aw1[e] = (k < 9) ? w1v : (_Float16)0.0f;
        }
        // per-lane channel biases: channel = v + 8*hl (+16 for second tile)
        float bb0[8], bb1[8];
        #pragma unroll
        for (int v = 0; v < 8; ++v) {
            bb0[v] = s_b1[8 * hl + v];
            bb1[v] = s_b1[16 + 8 * hl + v];
        }

        for (int t = wv; t < 43; t += 8) {            // 43 tiles cover 676 positions
            int p  = t * 16 + mn;                     // this lane's output pixel
            int pc = (p < 676) ? p : 675;
            int oy = pc / 26, ox = pc % 26;
            const _Float16* xb = &s_xh[oy * 28 + ox];
            v16h bx;                                  // B = im2col column for pos p
            #pragma unroll
            for (int e = 0; e < 16; ++e) {
                int k  = kmapA(e, hl);                // same K order as A; k>=9
                int kc = (k < 9) ? k : 8;             // is don't-care (A==0 there)
                bx[e] = xb[(kc / 3) * 28 + (kc % 3)];
            }
            v8f acc0 = {}, acc1 = {};
            acc0 = __builtin_amdgcn_wmma_f32_16x16x32_f16(false, aw0, false, bx, (short)0, acc0, false, false);
            acc1 = __builtin_amdgcn_wmma_f32_16x16x32_f16(false, aw1, false, bx, (short)0, acc1, false, false);
            if (p < 676) {                            // single per-lane guard
                v8h r0, r1v;
                #pragma unroll
                for (int v = 0; v < 8; ++v) {
                    r0[v]  = (_Float16)fmaxf(acc0[v] + bb0[v], 0.0f);
                    r1v[v] = (_Float16)fmaxf(acc1[v] + bb1[v], 0.0f);
                }
                *(v8h*)&s_r1[p * 32 + hl * 8]      = r0;   // channels 8*hl..8*hl+7
                *(v8h*)&s_r1[p * 32 + 16 + hl * 8] = r1v;  // channels 16+8*hl..
            }
        }
    }
    __syncthreads();

    // ---- stage 2: conv2 via WMMA (A frags preloaded, WMMAs back-to-back) -
    {
        const int mnw = (mn < 3) ? mn : 2;            // clamped, always-safe loads
        v16h bf[9];
        #pragma unroll
        for (int tap = 0; tap < 9; ++tap) {
            v16h b;
            #pragma unroll
            for (int e = 0; e < 16; ++e) {
                int ci = kmapA(e, hl);
                _Float16 w = s_w2h[(mnw * 32 + ci) * 9 + tap];
                b[e] = (mn < 3) ? w : (_Float16)0.0f; // register select
            }
            bf[tap] = b;
        }

        for (int t = wv; t < 36; t += 8) {            // 36 tiles of 16 positions
            int p  = t * 16 + mn;
            int oy = p / 24, ox = p % 24;
            const _Float16* base = &s_r1[(oy * 26 + ox) * 32 + hl * 8];
            v16h af[9];
            #pragma unroll
            for (int tap = 0; tap < 9; ++tap) {       // 18 ds_load_b128, one wait
                int off = ((tap / 3) * 26 + (tap % 3)) * 32;
                v8h lo = *(const v8h*)(base + off);
                v8h hi = *(const v8h*)(base + off + 16);
                af[tap] = __builtin_shufflevector(lo, hi,
                              0,1,2,3,4,5,6,7,8,9,10,11,12,13,14,15);
            }
            v8f acc = {};
            #pragma unroll
            for (int tap = 0; tap < 9; ++tap)
                acc = __builtin_amdgcn_wmma_f32_16x16x32_f16(
                          false, af[tap], false, bf[tap], (short)0, acc, false, false);
            if (mn < 3) {
                float bb = s_b2[mn];
                #pragma unroll
                for (int v = 0; v < 8; ++v) {
                    int row = t * 16 + v + 8 * hl;
                    s_c2[mn * 576 + row] = acc[v] + bb;
                }
            }
        }
    }
    __syncthreads();

    // ---- stage 3a: SIFT partials, one thread per (patch, cell, row-pair) -
    if (tid < 144) {
        int c    = tid / 48;
        int rem  = tid % 48;
        int cell = rem / 3;
        int seg  = rem % 3;                           // rows seg*2, seg*2+1 of the cell
        int cr = cell >> 2, cc = cell & 3;
        const float* P = &s_c2[c * 576];
        float hloc[8];
        #pragma unroll
        for (int k = 0; k < 8; ++k) hloc[k] = 0.0f;
        for (int iy = seg * 2; iy < seg * 2 + 2; ++iy) {
            for (int ix = 0; ix < 6; ++ix) {
                int h = cr * 6 + iy, w = cc * 6 + ix;
                float p00 = (h > 0  && w > 0 ) ? P[(h-1)*24 + (w-1)] : 0.f;
                float p01 = (h > 0           ) ? P[(h-1)*24 +  w   ] : 0.f;
                float p02 = (h > 0  && w < 23) ? P[(h-1)*24 + (w+1)] : 0.f;
                float p10 = (           w > 0 ) ? P[ h   *24 + (w-1)] : 0.f;
                float p12 = (           w < 23) ? P[ h   *24 + (w+1)] : 0.f;
                float p20 = (h < 23 && w > 0 ) ? P[(h+1)*24 + (w-1)] : 0.f;
                float p21 = (h < 23          ) ? P[(h+1)*24 +  w   ] : 0.f;
                float p22 = (h < 23 && w < 23) ? P[(h+1)*24 + (w+1)] : 0.f;
                float Ix = (p02 - p00) + 2.f * (p12 - p10) + (p22 - p20);
                float Iy = (p20 - p00) + 2.f * (p21 - p01) + (p22 - p02);
                float mag = sqrtf(Ix * Ix + Iy * Iy + 1e-12f);
                float ck[8]; float best = -INFINITY;
                #pragma unroll
                for (int k = 0; k < 8; ++k) {
                    ck[k] = s_trig[k] * Ix + s_trig[8 + k] * Iy;
                    best = fmaxf(best, ck[k]);
                }
                #pragma unroll
                for (int k = 0; k < 8; ++k)
                    if (ck[k] == best) hloc[k] += mag;   // ties: all maxima counted
            }
        }
        #pragma unroll
        for (int k = 0; k < 8; ++k)
            s_part[((c * 16 + cell) * 3 + seg) * 8 + k] = hloc[k];
    }
    __syncthreads();

    // ---- stage 3b: fixed-order combine -> hist (deterministic) ----------
    if (tid < 48) {
        int c = tid >> 4, cell = tid & 15;
        const float* pp = &s_part[(c * 16 + cell) * 3 * 8];
        #pragma unroll
        for (int k = 0; k < 8; ++k)
            s_hist[c * 128 + k * 16 + cell] = (pp[k] + pp[8 + k]) + pp[16 + k];
    }
    __syncthreads();

    // ---- L2 norm per patch (one wave each) ------------------------------
    if (wv < 3) {
        float ss = 0.f;
        for (int j = lane; j < 128; j += 32) {
            float f = s_hist[wv * 128 + j];
            ss += f * f;
        }
        #pragma unroll
        for (int off = 16; off >= 1; off >>= 1) ss += __shfl_xor(ss, off, 32);
        if (lane == 0) s_scale[wv] = 1.0f / (sqrtf(ss) + 1e-8f);
    }
    __syncthreads();

    for (int i = tid; i < 384; i += 256) {
        int c = i >> 7;
        float f = s_hist[i] * s_scale[c];
        feats[img * 384 + i] = (_Float16)powf(f + 1e-8f, 0.9f);
    }
}

// ---------------------------------------------------------------------------
// Kernel 2: logits = feats @ Wl^T + bl, relu, softmax.  One wave per 16 rows,
// 12x v_wmma_f32_16x16x32_f16 (K=384); Wl staged once per block as f16 in LDS;
// softmax via shfl over the 16-lane column groups of the C/D layout.
// ---------------------------------------------------------------------------
__global__ __launch_bounds__(256)
void classifier_softmax_kernel(const _Float16* __restrict__ feats,
                               const float* __restrict__ Wl,
                               const float* __restrict__ bl,
                               float* __restrict__ out)
{
    __shared__ __align__(16) _Float16 s_wl[10 * 384];   // 7680 B
    __shared__ float s_bl[10];

    const int tid  = threadIdx.x;
    const int lane = tid & 31;
    const int wv   = tid >> 5;
    const int tile = blockIdx.x * 8 + wv;            // 512 tiles total
    const int mn   = lane & 15;                      // A-row offset AND D-column
    const int hl   = lane >> 4;
    const int rowbase = tile * 16;
    const int wrowidx = (mn < 10) ? mn : 0;          // keep LDS reads in-bounds

    for (int i = tid; i < 3840; i += 256) s_wl[i] = (_Float16)Wl[i];
    if (tid < 10) s_bl[tid] = bl[tid];
    __syncthreads();

    v8f acc = {};
    #pragma unroll
    for (int ks = 0; ks < 12; ++ks) {
        const _Float16* arow = feats + (rowbase + mn) * 384 + ks * 32;
        v8h lo = *(const v8h*)(arow + hl * 8);
        v8h hi = *(const v8h*)(arow + 16 + hl * 8);
        v16h a = __builtin_shufflevector(lo, hi,
                                         0,1,2,3,4,5,6,7,8,9,10,11,12,13,14,15);
        const _Float16* wrow = &s_wl[wrowidx * 384 + ks * 32 + hl * 8];
        v8h wlo = *(const v8h*)(wrow);
        v8h whi = *(const v8h*)(wrow + 16);
        v16h bv = __builtin_shufflevector(wlo, whi,
                                          0,1,2,3,4,5,6,7,8,9,10,11,12,13,14,15);
        v16h b = (mn < 10) ? bv : (v16h)(_Float16)0.0f;
        acc = __builtin_amdgcn_wmma_f32_16x16x32_f16(
                  false, a, false, b, (short)0, acc, false, false);
    }

    float bias = (mn < 10) ? s_bl[wrowidx] : 0.0f;
    #pragma unroll
    for (int v = 0; v < 8; ++v) {
        int r = rowbase + v + 8 * hl;                // global output row
        float s = fmaxf(acc[v] + bias, 0.0f);        // relu(logit)
        float t = (mn < 10) ? s : -INFINITY;
        #pragma unroll
        for (int off = 8; off >= 1; off >>= 1) t = fmaxf(t, __shfl_xor(t, off, 32));
        float e = (mn < 10) ? __expf(s - t) : 0.0f;
        float sum = e;
        #pragma unroll
        for (int off = 8; off >= 1; off >>= 1) sum += __shfl_xor(sum, off, 32);
        if (mn < 10) out[r * 10 + mn] = e / sum;
    }
}

// ---------------------------------------------------------------------------
extern "C" void kernel_launch(void* const* d_in, const int* in_sizes, int n_in,
                              void* d_out, int out_size, void* d_ws, size_t ws_size,
                              hipStream_t stream)
{
    (void)in_sizes; (void)n_in; (void)out_size; (void)ws_size;
    const float* x  = (const float*)d_in[0];
    const float* W1 = (const float*)d_in[1];
    const float* b1 = (const float*)d_in[2];
    const float* W2 = (const float*)d_in[3];
    const float* b2 = (const float*)d_in[4];
    const float* Wl = (const float*)d_in[5];
    const float* bl = (const float*)d_in[6];
    float* out = (float*)d_out;

    _Float16* feats = (_Float16*)d_ws;               // 8192*384 f16 = 6.3 MB

    fused_conv_sift_kernel<<<8192, 256, 0, stream>>>(x, W1, b1, W2, b2, feats);
    classifier_softmax_kernel<<<64, 256, 0, stream>>>(feats, Wl, bl, out);
}
